// MultiheadAttention_28149215658101
// MI455X (gfx1250) — compile-verified, matrix operand order checked by reading, not run on hardware
//
#include <hip/hip_runtime.h>
#include <hip/hip_bf16.h>

// ---------------- CDNA5 WMMA / TDM types ----------------
typedef __attribute__((ext_vector_type(16))) __bf16       v16bf;
typedef __attribute__((ext_vector_type(8)))  float        v8f;
typedef __attribute__((ext_vector_type(4)))  unsigned int v4u;
typedef __attribute__((ext_vector_type(8)))  int          v8i;
typedef __attribute__((ext_vector_type(4)))  int          v4i;

struct alignas(16) Chunk128 { unsigned long long lo, hi; };
union FragU { Chunk128 c[2]; v16bf v; };

__device__ __forceinline__ v8f wmma_bf16(v16bf a, v16bf b, v8f acc) {
    // D = A(16x32) * B(32x16) + C, f32 accumulate
    return __builtin_amdgcn_wmma_f32_16x16x32_bf16(
        false, a, false, b, (short)0, acc, false, false);
}

// Load a 16x32 bf16 fragment (A-layout; also used for B stored (N,K) row-major).
// Per ISA 7.12.2: lane holds row (lane&15); lanes 0-15 carry K = k0+0..7 and
// k0+16..23, lanes 16-31 carry K = k0+8..15 and k0+24..31 -> two 16B chunks.
__device__ __forceinline__ v16bf load_frag(const __bf16* p, int row, int ld,
                                           int k0, int half) {
    const __bf16* base = p + (size_t)row * ld + k0 + 8 * half;
    FragU f;
    f.c[0] = *(const Chunk128*)(base);
    f.c[1] = *(const Chunk128*)(base + 16);
    return f.v;
}

// Max-reduce across each 16-lane row using DPP row-rotates (VALU only, no LDS).
// dpp_ctrl must be a compile-time constant -> template parameter.
template <int CTRL>
__device__ __forceinline__ float dpp_max_step(float v) {
    int s = __builtin_amdgcn_update_dpp(0, __float_as_int(v), CTRL, 0xf, 0xf,
                                        true);
    return fmaxf(v, __int_as_float(s));
}

__device__ __forceinline__ float rmax16_dpp(float v) {
    v = dpp_max_step<0x121>(v);  // row_ror:1
    v = dpp_max_step<0x122>(v);  // row_ror:2
    v = dpp_max_step<0x124>(v);  // row_ror:4
    v = dpp_max_step<0x128>(v);  // row_ror:8
    return v;
}

__device__ __forceinline__ v16bf ones_frag() {
    v16bf o;
#pragma unroll
    for (int i = 0; i < 16; i++) o[i] = (__bf16)1.0f;
    return o;
}

// ---------------- Tensor Data Mover: 2D tile -> LDS ----------------
// Pads LDS rows: 16 DWORDs (32 bf16) of data + 4 DWORDs pad -> ld = 40 halves.
__device__ __forceinline__ void tdm_load_2d(unsigned lds_off, const void* gptr,
                                            int tensor_d0, int tensor_d1,
                                            int stride_d0, int tile_d0,
                                            int tile_d1) {
    unsigned long long ga = (unsigned long long)gptr;
    v4u g0;
    g0[0] = 1u;                                  // count=1 valid descriptor
    g0[1] = lds_off;                             // LDS byte address
    g0[2] = (unsigned)(ga & 0xffffffffu);        // global_addr[31:0]
    g0[3] = (unsigned)((ga >> 32) & 0x01ffffffu) // global_addr[56:32]
            | (2u << 30);                        // type = 2 ("image")
    v8i g1;
    g1[0] = (1 << 16)        // data_size = 1 -> 2 bytes
          | (1 << 20)        // pad_enable
          | (3 << 22)        // pad_interval: 16 DWORDs between pads
          | (3 << 25);       // pad_amount:   4 DWORDs of pad
    g1[1] = (tensor_d0 & 0xffff) << 16;                       // tensor_dim0 lo
    g1[2] = ((unsigned)tensor_d0 >> 16) | ((tensor_d1 & 0xffff) << 16);
    g1[3] = ((unsigned)tensor_d1 >> 16) | ((tile_d0 & 0xffff) << 16);
    g1[4] = (tile_d1 & 0xffff);                               // tile_dim1, dim2=0
    g1[5] = stride_d0;                                        // dim0 stride lo
    g1[6] = 0;
    g1[7] = 0;
    v4i z4 = {0, 0, 0, 0};
#if __has_include(<hip/amd_detail/amd_gfx1250_TDM.h>)
    v8i z8 = {0, 0, 0, 0, 0, 0, 0, 0};
    __builtin_amdgcn_tensor_load_to_lds(g0, g1, z4, z4, z8, 0);
#else
    __builtin_amdgcn_tensor_load_to_lds(g0, g1, z4, z4, 0);
#endif
}

// ---------------- constants ----------------
#define SEQ  2048
#define BAT  2
#define DIM  1024
#define NH   16
#define DH   64
#define MROWS (SEQ * BAT)   // 4096

// ---------------- fp32 -> bf16 convert ----------------
__global__ void cvt_bf16_kernel(const float* __restrict__ src,
                                __bf16* __restrict__ dst, int n) {
    int i = blockIdx.x * blockDim.x + threadIdx.x;
    if (i < n) dst[i] = (__bf16)src[i];
}

// ---------------- TDM-staged WMMA GEMM: C = A(M,K) * W(N,K)^T + bias --------
// MODE 0: out = bf16, permuted to [B,H,S,DH], scaled (Q gets 1/sqrt(DH))
// MODE 1: out = fp32 [M,N]
template <int MODE>
__global__ __launch_bounds__(256, 1)
void gemm_bf16_kernel(const __bf16* __restrict__ A, const __bf16* __restrict__ W,
                      const float* __restrict__ bias, void* __restrict__ out,
                      float scale) {
    constexpr int LDT = 40;  // 32 data + 8 pad halves (TDM pad reproduces this)
    __shared__ __align__(16) __bf16 As[128 * LDT];
    __shared__ __align__(16) __bf16 Bs[64 * LDT];

    const int t = threadIdx.x;
    const int lane = t & 31, wave = t >> 5;
    const int lanelo = lane & 15, half = lane >> 4;
    const int wm = wave >> 1, wn = wave & 1;          // 4x2 wave grid
    const int m0 = blockIdx.x * 128;                  // M tile
    const int n0 = blockIdx.y * 64;                   // N tile

    const unsigned as_off = (unsigned)(unsigned long long)(void*)As;
    const unsigned bs_off = (unsigned)(unsigned long long)(void*)Bs;

    v8f acc[2][2] = {};

    for (int k0 = 0; k0 < DIM; k0 += 32) {
        __syncthreads();  // previous tile's fragment reads complete
        if (wave == 0) {
            // async DMA: A tile 32x128 rows, W tile 32x64 rows (K contiguous)
            tdm_load_2d(as_off, A + (size_t)m0 * DIM + k0, DIM, MROWS, DIM,
                        32, 128);
            tdm_load_2d(bs_off, W + (size_t)n0 * DIM + k0, DIM, DIM, DIM,
                        32, 64);
            __builtin_amdgcn_s_wait_tensorcnt(0);
        }
        __syncthreads();  // tiles visible to all waves

        v16bf af[2], bf[2];
#pragma unroll
        for (int i = 0; i < 2; i++)
            af[i] = load_frag(As, wm * 32 + i * 16 + lanelo, LDT, 0, half);
#pragma unroll
        for (int j = 0; j < 2; j++)
            bf[j] = load_frag(Bs, wn * 32 + j * 16 + lanelo, LDT, 0, half);
#pragma unroll
        for (int i = 0; i < 2; i++)
#pragma unroll
            for (int j = 0; j < 2; j++)
                acc[i][j] = wmma_bf16(af[i], bf[j], acc[i][j]);
    }

    // epilogue: C layout -> row M = r + 8*half, col N = lanelo (per 16x16 tile)
#pragma unroll
    for (int i = 0; i < 2; i++) {
#pragma unroll
        for (int j = 0; j < 2; j++) {
#pragma unroll
            for (int r = 0; r < 8; r++) {
                int m = m0 + wm * 32 + i * 16 + r + 8 * half;
                int n = n0 + wn * 32 + j * 16 + lanelo;
                float v = acc[i][j][r] + bias[n];
                if (MODE == 0) {
                    v *= scale;
                    int s = m >> 1, bb = m & 1;           // m = s*B + b
                    int hh = n >> 6, dh = n & 63;         // n = h*DH + dh
                    ((__bf16*)out)[(((size_t)(bb * NH + hh) * SEQ) + s) * DH + dh] =
                        (__bf16)v;
                } else {
                    ((float*)out)[(size_t)m * DIM + n] = v;
                }
            }
        }
    }
}

// ---------------- streaming-softmax attention ----------------
// grid: (SEQ/64, H, B), block: 128 (4 waves, 16 queries/wave)
__global__ __launch_bounds__(128, 1)
void attn_kernel(const __bf16* __restrict__ Q, const __bf16* __restrict__ K,
                 const __bf16* __restrict__ V, __bf16* __restrict__ O) {
    constexpr int LDT = 72;  // padded (multiple of 8)
    __shared__ __align__(16) __bf16 vt[64 * LDT];        // V^T tile [dh][key]
    __shared__ __align__(16) __bf16 pls[4][16 * LDT];    // per-wave P tile

    const int b = blockIdx.z, h = blockIdx.y;
    const int t = threadIdx.x;
    const int lane = t & 31, wave = t >> 5;
    const int lanelo = lane & 15, half = lane >> 4;
    const int q0 = blockIdx.x * 64 + wave * 16;

    const __bf16* Qh = Q + ((size_t)(b * NH + h) * SEQ) * DH;
    const __bf16* Kh = K + ((size_t)(b * NH + h) * SEQ) * DH;
    const __bf16* Vh = V + ((size_t)(b * NH + h) * SEQ) * DH;

    const v16bf qa0 = load_frag(Qh, q0 + lanelo, DH, 0, half);
    const v16bf qa1 = load_frag(Qh, q0 + lanelo, DH, 32, half);
    const v16bf onesv = ones_frag();

    v8f o[4] = {};
    float mrow[8], lrow[8];
#pragma unroll
    for (int r = 0; r < 8; r++) { mrow[r] = -3.0e38f; lrow[r] = 0.0f; }

    for (int kt = 0; kt < SEQ; kt += 64) {
        __syncthreads();
        // cooperatively transpose V tile (64 keys x 64 dh) into LDS
        {
            int key = t >> 1, c0 = (t & 1) * 32;
            Chunk128 tmp[4];
            const Chunk128* src =
                (const Chunk128*)(Vh + (size_t)(kt + key) * DH + c0);
#pragma unroll
            for (int c = 0; c < 4; c++) tmp[c] = src[c];
            const __bf16* e = (const __bf16*)tmp;
#pragma unroll
            for (int c = 0; c < 32; c++) vt[(c0 + c) * LDT + key] = e[c];
        }
        __syncthreads();

        if (kt + 64 < SEQ) {  // speculative L2 prefetch of next K/V tiles
            __builtin_prefetch(Kh + (size_t)(kt + 64 + lane * 2) * DH, 0, 1);
            __builtin_prefetch(Vh + (size_t)(kt + 64 + lane * 2) * DH, 0, 1);
        }

        // scores S = Q * K^T  (4 x 16x16 tiles across 64 keys)
        v8f sa[4];
#pragma unroll
        for (int j = 0; j < 4; j++) {
            v16bf kb0 = load_frag(Kh, kt + j * 16 + lanelo, DH, 0, half);
            v16bf kb1 = load_frag(Kh, kt + j * 16 + lanelo, DH, 32, half);
            v8f c = {};
            c = wmma_bf16(qa0, kb0, c);
            c = wmma_bf16(qa1, kb1, c);
            sa[j] = c;
        }

        // online softmax: row max across 64 keys (DPP rotate-reduce, no LDS)
        float alpha[8];
#pragma unroll
        for (int r = 0; r < 8; r++) {
            float tm = sa[0][r];
#pragma unroll
            for (int j = 1; j < 4; j++) tm = fmaxf(tm, sa[j][r]);
            tm = rmax16_dpp(tm);
            float mn = fmaxf(mrow[r], tm);
            alpha[r] = __expf(mrow[r] - mn);
            mrow[r] = mn;
        }
        // P = exp(S - m) -> per-wave LDS (transposed into A layout)
#pragma unroll
        for (int j = 0; j < 4; j++) {
#pragma unroll
            for (int r = 0; r < 8; r++) {
                float p = __expf(sa[j][r] - mrow[r]);
                pls[wave][(r + 8 * half) * LDT + j * 16 + lanelo] = (__bf16)p;
            }
        }

        v16bf pa0 = load_frag(pls[wave], lanelo, LDT, 0, half);
        v16bf pa1 = load_frag(pls[wave], lanelo, LDT, 32, half);

        // row sums via ones-matrix WMMA: every lane gets rowsum in element r
        v8f lt = {};
        lt = wmma_bf16(pa0, onesv, lt);
        lt = wmma_bf16(pa1, onesv, lt);
#pragma unroll
        for (int r = 0; r < 8; r++) {
            lrow[r] = lrow[r] * alpha[r] + lt[r];
#pragma unroll
            for (int j = 0; j < 4; j++) o[j][r] *= alpha[r];
        }

        // O += P(16x64) * V(64x64)
#pragma unroll
        for (int jn = 0; jn < 4; jn++) {
            v16bf vb0 = load_frag(vt, jn * 16 + lanelo, LDT, 0, half);
            v16bf vb1 = load_frag(vt, jn * 16 + lanelo, LDT, 32, half);
            o[jn] = wmma_bf16(pa0, vb0, o[jn]);
            o[jn] = wmma_bf16(pa1, vb1, o[jn]);
        }
    }

    // normalize and write O back as [M = s*B+b, D] bf16 for the final GEMM
#pragma unroll
    for (int r = 0; r < 8; r++) {
        float inv = 1.0f / lrow[r];
        int q = q0 + r + 8 * half;
        size_t m = (size_t)q * BAT + b;
#pragma unroll
        for (int j = 0; j < 4; j++) {
            int d = h * DH + j * 16 + lanelo;
            O[m * DIM + d] = (__bf16)(o[j][r] * inv);
        }
    }
}

// ---------------- launcher ----------------
extern "C" void kernel_launch(void* const* d_in, const int* in_sizes, int n_in,
                              void* d_out, int out_size, void* d_ws, size_t ws_size,
                              hipStream_t stream) {
    const float* x  = (const float*)d_in[0];
    const float* Wq = (const float*)d_in[1];
    const float* bq = (const float*)d_in[2];
    const float* Wk = (const float*)d_in[3];
    const float* bk = (const float*)d_in[4];
    const float* Wv = (const float*)d_in[5];
    const float* bv = (const float*)d_in[6];
    const float* Wo = (const float*)d_in[7];
    const float* bo = (const float*)d_in[8];

    char* ws = (char*)d_ws;                      // ~48 MiB used
    __bf16* xb  = (__bf16*)(ws);                 // [4096,1024]   8 MiB
    __bf16* wqb = (__bf16*)(ws + (8ull  << 20)); // [1024,1024]   2 MiB
    __bf16* wkb = (__bf16*)(ws + (10ull << 20));
    __bf16* wvb = (__bf16*)(ws + (12ull << 20));
    __bf16* wob = (__bf16*)(ws + (14ull << 20));
    __bf16* Qb  = (__bf16*)(ws + (16ull << 20)); // [B,H,S,DH]    8 MiB
    __bf16* Kb  = (__bf16*)(ws + (24ull << 20));
    __bf16* Vb  = (__bf16*)(ws + (32ull << 20));
    __bf16* Ob  = (__bf16*)(ws + (40ull << 20)); // [4096,1024]   8 MiB

    const int nX = MROWS * DIM, nW = DIM * DIM;
    cvt_bf16_kernel<<<(nX + 255) / 256, 256, 0, stream>>>(x,  xb,  nX);
    cvt_bf16_kernel<<<(nW + 255) / 256, 256, 0, stream>>>(Wq, wqb, nW);
    cvt_bf16_kernel<<<(nW + 255) / 256, 256, 0, stream>>>(Wk, wkb, nW);
    cvt_bf16_kernel<<<(nW + 255) / 256, 256, 0, stream>>>(Wv, wvb, nW);
    cvt_bf16_kernel<<<(nW + 255) / 256, 256, 0, stream>>>(Wo, wob, nW);

    dim3 ggrid(MROWS / 128, DIM / 64);
    const float qscale = 0.125f;  // 1/sqrt(DH)
    gemm_bf16_kernel<0><<<ggrid, 256, 0, stream>>>(xb, wqb, bq, Qb, qscale);
    gemm_bf16_kernel<0><<<ggrid, 256, 0, stream>>>(xb, wkb, bk, Kb, 1.0f);
    gemm_bf16_kernel<0><<<ggrid, 256, 0, stream>>>(xb, wvb, bv, Vb, 1.0f);

    attn_kernel<<<dim3(SEQ / 64, NH, BAT), 128, 0, stream>>>(Qb, Kb, Vb, Ob);

    gemm_bf16_kernel<1><<<ggrid, 256, 0, stream>>>(Ob, wob, bo, d_out, 1.0f);
}
